// DiffusionTransformerBlock_18124761989724
// MI455X (gfx1250) — compile-verified, matrix operand order checked by reading, not run on hardware
//
#include <hip/hip_runtime.h>

typedef __bf16 bf16;
typedef __attribute__((ext_vector_type(16))) __bf16 bf16x16;
typedef __attribute__((ext_vector_type(8)))  float  floatx8;

#define WMMA_BF16(A, B, C) \
  __builtin_amdgcn_wmma_f32_16x16x32_bf16(false, (A), false, (B), (short)0, (C), false, false)

// ---------------------------------------------------------------------------
// Contiguous WMMA fragment loader (CDNA5 16-bit layouts, wave32).
// Per-lane element->K mapping: e=0..7 -> k = half*8 + e ; e=8..15 -> k = 16 + half*8 + (e-8)
// => each lane's 16 bf16 are two contiguous 16B chunks at (row*ld + half*8) and (+16).
// Works for A (lane = row M) and for B when B is stored transposed [N][K]
// (lane = column N, elements = K), since both use lane%16 + the same K map.
// ---------------------------------------------------------------------------
struct f4x2 { float4 a, b; };

__device__ __forceinline__ bf16x16 load_frag_row(const bf16* base, int ld) {
  int lane = threadIdx.x & 31;
  int row = lane & 15, half = lane >> 4;
  const bf16* p = base + row * ld + half * 8;
  f4x2 t;
  t.a = *(const float4*)(p);
  t.b = *(const float4*)(p + 16);
  return __builtin_bit_cast(bf16x16, t);
}

__device__ __forceinline__ float dev_sigmoid(float v) {
  return __builtin_amdgcn_rcpf(1.0f + __expf(-v));
}

__device__ __forceinline__ unsigned pack_bf16(float a, float b) {
  unsigned short ua = __builtin_bit_cast(unsigned short, (bf16)a);
  unsigned short ub = __builtin_bit_cast(unsigned short, (bf16)b);
  return (unsigned)ua | ((unsigned)ub << 16);
}

// ---------------------------------------------------------------------------
// Generic bf16 WMMA GEMM: C[M,N] = epilogue(A[M,K] @ B[K,N]), B given as B^T [N,K].
// block = 256 threads = 8 waves; block tile 64x64; wave tile 16x32; K step 32.
// ACT: 0 none, 1 sigmoid, 2 silu.
// HP: 0 normal bf16 out; 1 bf16 out [h][row][64] (head-pad); 2 bf16 out [h][64][1024] (head-T)
// ---------------------------------------------------------------------------
template <bool HAS_BIAS, int ACT, bool HAS_MUL, bool HAS_OUTF, bool HAS_OUTB, int HP>
__global__ __launch_bounds__(256)
void gemm_bf16(const bf16* __restrict__ A, const bf16* __restrict__ BT,
               int M, int N, int K, const float* __restrict__ bias,
               float post_scale, const float* __restrict__ mulbuf,
               float* __restrict__ outF, bf16* __restrict__ outB) {
  __shared__ bf16 As[64][48];  // row stride 96B (16B aligned)
  __shared__ bf16 Bs[64][48];  // B^T tile: [n][k]

  int tid = threadIdx.x;
  int lane = tid & 31;
  int wave = tid >> 5;
  int wm = wave & 3;   // wave row 0..3
  int wn = wave >> 2;  // wave col 0..1
  int rowBase = blockIdx.y * 64;
  int colBase = blockIdx.x * 64;

  int r = tid >> 2, c8 = (tid & 3) * 8;  // 64x32 tile, 8 bf16 per thread

  floatx8 acc0 = {};
  floatx8 acc1 = {};

  for (int k0 = 0; k0 < K; k0 += 32) {
    float4 av = *(const float4*)(A + (size_t)(rowBase + r) * K + k0 + c8);
    float4 bv = *(const float4*)(BT + (size_t)(colBase + r) * K + k0 + c8);
    __syncthreads();
    *(float4*)(&As[r][c8]) = av;
    *(float4*)(&Bs[r][c8]) = bv;
    __syncthreads();
    bf16x16 af = load_frag_row(&As[wm * 16][0], 48);
    bf16x16 b0 = load_frag_row(&Bs[wn * 32][0], 48);
    bf16x16 b1 = load_frag_row(&Bs[wn * 32 + 16][0], 48);
    acc0 = WMMA_BF16(af, b0, acc0);
    acc1 = WMMA_BF16(af, b1, acc1);
  }

  int half = lane >> 4, cn = lane & 15;
#pragma unroll
  for (int j = 0; j < 8; ++j) {
    int grow = rowBase + wm * 16 + j + 8 * half;
#pragma unroll
    for (int sub = 0; sub < 2; ++sub) {
      int gcol = colBase + wn * 32 + sub * 16 + cn;
      float v = (sub == 0) ? acc0[j] : acc1[j];
      if constexpr (HAS_BIAS) v += bias[gcol];
      v *= post_scale;
      if constexpr (ACT == 1) v = dev_sigmoid(v);
      else if constexpr (ACT == 2) v = v * dev_sigmoid(v);
      size_t idx = (size_t)grow * N + gcol;
      if constexpr (HAS_MUL) v *= mulbuf[idx];
      if constexpr (HAS_OUTF) outF[idx] = v;
      if constexpr (HAS_OUTB) {
        if constexpr (HP == 1) {
          int h = gcol / 48, d = gcol % 48;
          outB[((size_t)h * M + grow) * 64 + d] = (bf16)v;
        } else if constexpr (HP == 2) {
          int h = gcol / 48, d = gcol % 48;
          outB[((size_t)h * 64 + d) * 1024 + grow] = (bf16)v;
        } else {
          outB[idx] = (bf16)v;
        }
      }
    }
  }
}

// ---------------------------------------------------------------------------
// Row LayerNorm (one block per row). Optional affine, fp32/bf16 outputs.
// ---------------------------------------------------------------------------
__global__ __launch_bounds__(256)
void row_ln(const float* __restrict__ in, int cols,
            const float* __restrict__ g, const float* __restrict__ b,
            float* __restrict__ outF, bf16* __restrict__ outB) {
  int row = blockIdx.x;
  const float* x = in + (size_t)row * cols;
  float s = 0.f, ss = 0.f;
  for (int c = threadIdx.x; c < cols; c += blockDim.x) {
    float v = x[c];
    s += v;
    ss += v * v;
  }
  __shared__ float r1[256], r2[256];
  r1[threadIdx.x] = s;
  r2[threadIdx.x] = ss;
  __syncthreads();
  for (int off = 128; off > 0; off >>= 1) {
    if ((int)threadIdx.x < off) {
      r1[threadIdx.x] += r1[threadIdx.x + off];
      r2[threadIdx.x] += r2[threadIdx.x + off];
    }
    __syncthreads();
  }
  float mean = r1[0] / (float)cols;
  float var  = r2[0] / (float)cols - mean * mean;
  float rs = rsqrtf(var + 1e-5f);
  for (int c = threadIdx.x; c < cols; c += blockDim.x) {
    float v = (x[c] - mean) * rs;
    if (g) v = v * g[c] + b[c];
    if (outF) outF[(size_t)row * cols + c] = v;
    if (outB) outB[(size_t)row * cols + c] = (bf16)v;
  }
}

// ---------------------------------------------------------------------------
// Pair bias: bias[h][q][k] = (LN(z[q,k,:]; zg,zb) @ Wz)[h]
// one wave per 16 consecutive (q,k) rows; K=128 -> 4 WMMAs; WzT is [16][128]
// ---------------------------------------------------------------------------
__global__ __launch_bounds__(32)
void zbias_kernel(const float* __restrict__ z, const float* __restrict__ zg,
                  const float* __restrict__ zb, const bf16* __restrict__ WzT,
                  float* __restrict__ bias) {
  __shared__ float zf[16][128];
  __shared__ bf16 zn[16][136];
  int R = blockIdx.x * 16;
  int q = R >> 10;
  int kb = R & 1023;
  int lane = threadIdx.x;
  int half = lane >> 4, l16 = lane & 15;

  const float* src = z + (((size_t)q * 1024 + kb + l16) * 128) + half * 64;
  float s = 0.f, ss = 0.f;
#pragma unroll
  for (int i = 0; i < 16; ++i) {
    float4 v4 = *(const float4*)(src + i * 4);
    *(float4*)(&zf[l16][half * 64 + i * 4]) = v4;
    s += v4.x + v4.y + v4.z + v4.w;
    ss += v4.x * v4.x + v4.y * v4.y + v4.z * v4.z + v4.w * v4.w;
  }
  s  += __shfl_xor(s, 16, 32);
  ss += __shfl_xor(ss, 16, 32);
  float mean = s * (1.f / 128.f);
  float var  = ss * (1.f / 128.f) - mean * mean;
  float rs = rsqrtf(var + 1e-5f);
  __syncthreads();
#pragma unroll
  for (int i = 0; i < 16; ++i) {
    int c = half * 64 + i * 4;
    float4 xv = *(const float4*)(&zf[l16][c]);
    float4 gv = *(const float4*)(zg + c);
    float4 bv = *(const float4*)(zb + c);
    float y0 = (xv.x - mean) * rs * gv.x + bv.x;
    float y1 = (xv.y - mean) * rs * gv.y + bv.y;
    float y2 = (xv.z - mean) * rs * gv.z + bv.z;
    float y3 = (xv.w - mean) * rs * gv.w + bv.w;
    uint2 pk;
    pk.x = pack_bf16(y0, y1);
    pk.y = pack_bf16(y2, y3);
    *(uint2*)(&zn[l16][c]) = pk;
  }
  __syncthreads();

  floatx8 acc = {};
#pragma unroll
  for (int c = 0; c < 4; ++c) {
    bf16x16 af  = load_frag_row(&zn[0][c * 32], 136);
    bf16x16 bfr = load_frag_row(WzT + c * 32, 128);  // lane = h col, elems = k (contig)
    acc = WMMA_BF16(af, bfr, acc);
  }
#pragma unroll
  for (int j = 0; j < 8; ++j) {
    int kk = kb + j + 8 * half;
    int hh = l16;
    bias[((size_t)hh * 1024 + q) * 1024 + kk] = acc[j];
  }
}

// ---------------------------------------------------------------------------
// Attention: one wave per (head, 16-query block). D padded 48->64 (zeros).
// q,k: [h][n][64]; v: transposed [h][64][1024]. scores in LDS (16x1024 fp32).
// ---------------------------------------------------------------------------
__global__ __launch_bounds__(32)
void attention_kernel(const bf16* __restrict__ qp, const bf16* __restrict__ kp,
                      const bf16* __restrict__ vt,
                      const float* __restrict__ bias, const float* __restrict__ gate,
                      bf16* __restrict__ attB) {
  __shared__ float sc[16][1024];  // exactly 64KB
  int h = blockIdx.y;
  int qb = blockIdx.x;
  int qbase = qb * 16;
  int lane = threadIdx.x;
  int half = lane >> 4, l16 = lane & 15;

  const bf16* Q = qp + ((size_t)h * 1024 + qbase) * 64;
  bf16x16 aQ0 = load_frag_row(Q, 64);
  bf16x16 aQ1 = load_frag_row(Q + 32, 64);

  const float* biasRow = bias + ((size_t)h * 1024 + qbase) * 1024;

  for (int kt = 0; kt < 64; ++kt) {
    const bf16* Kt = kp + ((size_t)h * 1024 + kt * 16) * 64;
    bf16x16 b0 = load_frag_row(Kt, 64);       // lane = key col, elems = d (contig)
    bf16x16 b1 = load_frag_row(Kt + 32, 64);
    floatx8 c = {};
    c = WMMA_BF16(aQ0, b0, c);
    c = WMMA_BF16(aQ1, b1, c);
#pragma unroll
    for (int j = 0; j < 8; ++j) {
      int qr = j + 8 * half;
      int kc = kt * 16 + l16;
      sc[qr][kc] = c[j] + biasRow[(size_t)qr * 1024 + kc];
    }
  }
  __syncthreads();

  // softmax: 2 lanes per row; each scans 512 columns (vectorized)
  float rcp;
  {
    int row = l16;
    float m = -3.0e38f;
    for (int c0 = half * 512; c0 < half * 512 + 512; c0 += 4) {
      float4 v4 = *(const float4*)(&sc[row][c0]);
      m = fmaxf(m, fmaxf(fmaxf(v4.x, v4.y), fmaxf(v4.z, v4.w)));
    }
    m = fmaxf(m, __shfl_xor(m, 16, 32));
    float sum = 0.f;
    for (int c0 = half * 512; c0 < half * 512 + 512; c0 += 4) {
      float4 v4 = *(const float4*)(&sc[row][c0]);
      v4.x = __expf(v4.x - m);
      v4.y = __expf(v4.y - m);
      v4.z = __expf(v4.z - m);
      v4.w = __expf(v4.w - m);
      *(float4*)(&sc[row][c0]) = v4;
      sum += v4.x + v4.y + v4.z + v4.w;
    }
    sum += __shfl_xor(sum, 16, 32);
    rcp = __builtin_amdgcn_rcpf(sum);
  }
  __syncthreads();

  // P @ V  (only d<48 needed -> 3 N-tiles); V transposed so B frags contiguous
  floatx8 o0 = {}, o1 = {}, o2 = {};
  for (int ks = 0; ks < 32; ++ks) {
    const float* pr = &sc[l16][ks * 32];
    float4 f0 = *(const float4*)(pr + half * 8);
    float4 f1 = *(const float4*)(pr + half * 8 + 4);
    float4 f2 = *(const float4*)(pr + 16 + half * 8);
    float4 f3 = *(const float4*)(pr + 16 + half * 8 + 4);
    bf16x16 ap;
    ap[0]  = (bf16)(f0.x * rcp); ap[1]  = (bf16)(f0.y * rcp);
    ap[2]  = (bf16)(f0.z * rcp); ap[3]  = (bf16)(f0.w * rcp);
    ap[4]  = (bf16)(f1.x * rcp); ap[5]  = (bf16)(f1.y * rcp);
    ap[6]  = (bf16)(f1.z * rcp); ap[7]  = (bf16)(f1.w * rcp);
    ap[8]  = (bf16)(f2.x * rcp); ap[9]  = (bf16)(f2.y * rcp);
    ap[10] = (bf16)(f2.z * rcp); ap[11] = (bf16)(f2.w * rcp);
    ap[12] = (bf16)(f3.x * rcp); ap[13] = (bf16)(f3.y * rcp);
    ap[14] = (bf16)(f3.z * rcp); ap[15] = (bf16)(f3.w * rcp);

    const bf16* Vb = vt + (size_t)h * 64 * 1024 + ks * 32;
    bf16x16 b0 = load_frag_row(Vb,               1024);
    bf16x16 b1 = load_frag_row(Vb + 16 * 1024,   1024);
    bf16x16 b2 = load_frag_row(Vb + 32 * 1024,   1024);
    o0 = WMMA_BF16(ap, b0, o0);
    o1 = WMMA_BF16(ap, b1, o1);
    o2 = WMMA_BF16(ap, b2, o2);
  }

#pragma unroll
  for (int t = 0; t < 3; ++t) {
#pragma unroll
    for (int j = 0; j < 8; ++j) {
      int d = t * 16 + l16;
      if (d < 48) {
        int grow = qbase + j + 8 * half;
        int col = h * 48 + d;
        float v = (t == 0 ? o0[j] : (t == 1 ? o1[j] : o2[j]));
        v *= gate[(size_t)grow * 768 + col];
        attB[(size_t)grow * 768 + col] = (bf16)v;
      }
    }
  }
}

// ---------------------------------------------------------------------------
// Elementwise helpers
// ---------------------------------------------------------------------------
__global__ void cvt_f32_bf16(const float* __restrict__ in, bf16* __restrict__ out, int n) {
  int i = blockIdx.x * blockDim.x + threadIdx.x;
  if (i < n) out[i] = (bf16)in[i];
}

// fp32 [K,N] -> bf16 transposed [N,K]
__global__ void cvt_f32_bf16_t(const float* __restrict__ in, bf16* __restrict__ out,
                               int K, int N) {
  int i = blockIdx.x * blockDim.x + threadIdx.x;
  if (i < K * N) {
    int k = i / N, n = i % N;
    out[(size_t)n * K + k] = (bf16)in[i];
  }
}

__global__ void zero_bf16(bf16* __restrict__ out, int n) {
  int i = blockIdx.x * blockDim.x + threadIdx.x;
  if (i < n) out[i] = (bf16)0.0f;
}

// out = G*aln + Bm  (G already sigmoided by GEMM epilogue); bf16 out
__global__ void adaln_combine(const float* __restrict__ G, const float* __restrict__ Bm,
                              const float* __restrict__ aln, bf16* __restrict__ out, int n) {
  int i = blockIdx.x * blockDim.x + threadIdx.x;
  if (i < n) out[i] = (bf16)(G[i] * aln[i] + Bm[i]);
}

// out = gate*val + resid   (fp32)
__global__ void fma_res(const float* __restrict__ gate, const float* __restrict__ val,
                        const float* __restrict__ resid, float* __restrict__ out, int n) {
  int i = blockIdx.x * blockDim.x + threadIdx.x;
  if (i < n) out[i] = gate[i] * val[i] + resid[i];
}

// ---------------------------------------------------------------------------
// Host driver
// ---------------------------------------------------------------------------
extern "C" void kernel_launch(void* const* d_in, const int* in_sizes, int n_in,
                              void* d_out, int out_size, void* d_ws, size_t ws_size,
                              hipStream_t stream) {
  (void)in_sizes; (void)n_in; (void)out_size; (void)ws_size;

  const int M = 1024, CA = 768, CS = 384, FF = 1536;
  const int AE = M * CA;
  const int SE = M * CS;
  const int QKV = 16 * 1024 * 64;

  const float* a   = (const float*)d_in[0];
  const float* s   = (const float*)d_in[1];
  const float* z   = (const float*)d_in[2];
  const float* a1g = (const float*)d_in[3];
  const float* a1b = (const float*)d_in[4];
  const float* a1sgW = (const float*)d_in[5];
  const float* a1sgb = (const float*)d_in[6];
  const float* a1sbW = (const float*)d_in[7];
  const float* Wq = (const float*)d_in[8];
  const float* bq = (const float*)d_in[9];
  const float* Wk = (const float*)d_in[10];
  const float* Wv = (const float*)d_in[11];
  const float* Wg = (const float*)d_in[12];
  const float* Wo = (const float*)d_in[13];
  const float* bo = (const float*)d_in[14];
  const float* zlg = (const float*)d_in[15];
  const float* zlb = (const float*)d_in[16];
  const float* Wz = (const float*)d_in[17];
  const float* Wlast = (const float*)d_in[18];
  const float* blast = (const float*)d_in[19];
  const float* a2g = (const float*)d_in[20];
  const float* a2b = (const float*)d_in[21];
  const float* a2sgW = (const float*)d_in[22];
  const float* a2sgb = (const float*)d_in[23];
  const float* a2sbW = (const float*)d_in[24];
  const float* W1 = (const float*)d_in[25];
  const float* W2 = (const float*)d_in[26];
  const float* Wb = (const float*)d_in[27];
  const float* Ws = (const float*)d_in[28];
  const float* bs = (const float*)d_in[29];

  char* wsp = (char*)d_ws;
  size_t off = 0;
  auto alloc = [&](size_t bytes) -> void* {
    void* p = wsp + off;
    off += (bytes + 255) & ~(size_t)255;
    return p;
  };

  // bf16 weights (all stored TRANSPOSED: [N][K])
  bf16* w_a1sg = (bf16*)alloc(CS * CA * 2);
  bf16* w_a1sb = (bf16*)alloc(CS * CA * 2);
  bf16* w_q = (bf16*)alloc(CA * CA * 2);
  bf16* w_k = (bf16*)alloc(CA * CA * 2);
  bf16* w_v = (bf16*)alloc(CA * CA * 2);
  bf16* w_g = (bf16*)alloc(CA * CA * 2);
  bf16* w_o = (bf16*)alloc(CA * CA * 2);
  bf16* w_z = (bf16*)alloc(128 * 16 * 2);
  bf16* w_last = (bf16*)alloc(CS * CA * 2);
  bf16* w_a2sg = (bf16*)alloc(CS * CA * 2);
  bf16* w_a2sb = (bf16*)alloc(CS * CA * 2);
  bf16* w_1 = (bf16*)alloc(CA * FF * 2);
  bf16* w_2 = (bf16*)alloc(CA * FF * 2);
  bf16* w_b = (bf16*)alloc(FF * CA * 2);
  bf16* w_s = (bf16*)alloc(CS * CA * 2);
  // bf16 activations
  bf16* s_b    = (bf16*)alloc(SE * 2);
  bf16* sln1_b = (bf16*)alloc(SE * 2);
  bf16* sln2_b = (bf16*)alloc(SE * 2);
  bf16* x_b    = (bf16*)alloc(AE * 2);   // reused as t_b
  bf16* att_b  = (bf16*)alloc(AE * 2);
  bf16* hb_b   = (bf16*)alloc(M * FF * 2);
  bf16* qkvpad = (bf16*)alloc((size_t)3 * QKV * 2);
  bf16* qpad = qkvpad;            // [h][n][64]
  bf16* kpad = qkvpad + QKV;      // [h][n][64]
  bf16* vtr  = qkvpad + 2 * QKV;  // [h][64][1024]
  // fp32 buffers
  float* alnF  = (float*)alloc(AE * 4);
  float* Gf    = (float*)alloc(AE * 4);
  float* Bf    = (float*)alloc(AE * 4);
  float* gateF = (float*)alloc(AE * 4);
  float* projF = (float*)alloc(AE * 4);
  float* aoutF = (float*)alloc(AE * 4);
  float* h1F   = (float*)alloc((size_t)M * FF * 4);
  float* biasF = (float*)alloc((size_t)16 * 1024 * 1024 * 4);

  auto cvt = [&](const float* src, bf16* dst, int n) {
    cvt_f32_bf16<<<dim3((n + 255) / 256), dim3(256), 0, stream>>>(src, dst, n);
  };
  auto cvtT = [&](const float* src, bf16* dst, int K, int N) {
    int n = K * N;
    cvt_f32_bf16_t<<<dim3((n + 255) / 256), dim3(256), 0, stream>>>(src, dst, K, N);
  };

  // --- weight conversions (transposed) + s copy ---
  cvtT(a1sgW, w_a1sg, CS, CA);
  cvtT(a1sbW, w_a1sb, CS, CA);
  cvtT(Wq, w_q, CA, CA);
  cvtT(Wk, w_k, CA, CA);
  cvtT(Wv, w_v, CA, CA);
  cvtT(Wg, w_g, CA, CA);
  cvtT(Wo, w_o, CA, CA);
  cvtT(Wz, w_z, 128, 16);
  cvtT(Wlast, w_last, CS, CA);
  cvtT(a2sgW, w_a2sg, CS, CA);
  cvtT(a2sbW, w_a2sb, CS, CA);
  cvtT(W1, w_1, CA, FF);
  cvtT(W2, w_2, CA, FF);
  cvtT(Wb, w_b, FF, CA);
  cvtT(Ws, w_s, CS, CA);
  cvt(s, s_b, SE);
  zero_bf16<<<dim3((3 * QKV + 255) / 256), dim3(256), 0, stream>>>(qkvpad, 3 * QKV);

  dim3 blk(256);
  dim3 gA(CA / 64, M / 64);   // N=768
  dim3 gF(FF / 64, M / 64);   // N=1536

  // --- adaLN 1 ---
  row_ln<<<dim3(M), blk, 0, stream>>>(s, CS, a1g, a1b, nullptr, sln1_b);
  row_ln<<<dim3(M), blk, 0, stream>>>(a, CA, nullptr, nullptr, alnF, nullptr);
  gemm_bf16<true, 1, false, true, false, 0><<<gA, blk, 0, stream>>>(
      sln1_b, w_a1sg, M, CA, CS, a1sgb, 1.f, nullptr, Gf, nullptr);
  gemm_bf16<false, 0, false, true, false, 0><<<gA, blk, 0, stream>>>(
      sln1_b, w_a1sb, M, CA, CS, nullptr, 1.f, nullptr, Bf, nullptr);
  adaln_combine<<<dim3((AE + 255) / 256), blk, 0, stream>>>(Gf, Bf, alnF, x_b, AE);

  // --- projections ---
  const float qscale = 0.14433756729740643f;  // 48^-0.5
  gemm_bf16<true, 0, false, false, true, 1><<<gA, blk, 0, stream>>>(
      x_b, w_q, M, CA, CA, bq, qscale, nullptr, nullptr, qpad);
  gemm_bf16<false, 0, false, false, true, 1><<<gA, blk, 0, stream>>>(
      x_b, w_k, M, CA, CA, nullptr, 1.f, nullptr, nullptr, kpad);
  gemm_bf16<false, 0, false, false, true, 2><<<gA, blk, 0, stream>>>(
      x_b, w_v, M, CA, CA, nullptr, 1.f, nullptr, nullptr, vtr);
  gemm_bf16<false, 1, false, true, false, 0><<<gA, blk, 0, stream>>>(
      x_b, w_g, M, CA, CA, nullptr, 1.f, nullptr, gateF, nullptr);

  // --- pair bias from z ---
  zbias_kernel<<<dim3(1024 * 1024 / 16), dim3(32), 0, stream>>>(z, zlg, zlb, w_z, biasF);

  // --- attention ---
  attention_kernel<<<dim3(64, 16), dim3(32), 0, stream>>>(qpad, kpad, vtr, biasF, gateF, att_b);

  // --- output projection + s-gate + residual ---
  gemm_bf16<true, 0, false, true, false, 0><<<gA, blk, 0, stream>>>(
      att_b, w_o, M, CA, CA, bo, 1.f, nullptr, projF, nullptr);
  gemm_bf16<true, 1, false, true, false, 0><<<gA, blk, 0, stream>>>(
      s_b, w_last, M, CA, CS, blast, 1.f, nullptr, gateF, nullptr);
  fma_res<<<dim3((AE + 255) / 256), blk, 0, stream>>>(gateF, projF, a, aoutF, AE);

  // --- adaLN 2 ---
  row_ln<<<dim3(M), blk, 0, stream>>>(aoutF, CA, nullptr, nullptr, alnF, nullptr);
  row_ln<<<dim3(M), blk, 0, stream>>>(s, CS, a2g, a2b, nullptr, sln2_b);
  gemm_bf16<true, 1, false, true, false, 0><<<gA, blk, 0, stream>>>(
      sln2_b, w_a2sg, M, CA, CS, a2sgb, 1.f, nullptr, Gf, nullptr);
  gemm_bf16<false, 0, false, true, false, 0><<<gA, blk, 0, stream>>>(
      sln2_b, w_a2sb, M, CA, CS, nullptr, 1.f, nullptr, Bf, nullptr);
  adaln_combine<<<dim3((AE + 255) / 256), blk, 0, stream>>>(Gf, Bf, alnF, x_b, AE);

  // --- SwiGLU transition ---
  gemm_bf16<false, 2, false, true, false, 0><<<gF, blk, 0, stream>>>(
      x_b, w_1, M, FF, CA, nullptr, 1.f, nullptr, h1F, nullptr);
  gemm_bf16<false, 0, true, false, true, 0><<<gF, blk, 0, stream>>>(
      x_b, w_2, M, FF, CA, nullptr, 1.f, h1F, nullptr, hb_b);
  gemm_bf16<false, 0, false, true, false, 0><<<gA, blk, 0, stream>>>(
      hb_b, w_b, M, CA, FF, nullptr, 1.f, nullptr, projF, nullptr);
  gemm_bf16<true, 1, false, true, false, 0><<<gA, blk, 0, stream>>>(
      s_b, w_s, M, CA, CS, bs, 1.f, nullptr, gateF, nullptr);
  fma_res<<<dim3((AE + 255) / 256), blk, 0, stream>>>(gateF, projF, aoutF, (float*)d_out, AE);

  // --- pass-through outputs: s and z ---
  float* outp = (float*)d_out;
  hipMemcpyAsync(outp + AE, s, (size_t)SE * 4, hipMemcpyDeviceToDevice, stream);
  hipMemcpyAsync(outp + AE + SE, z, (size_t)1024 * 1024 * 128 * 4,
                 hipMemcpyDeviceToDevice, stream);
}
